// GCNNet_79216376808038
// MI455X (gfx1250) — matrix operand order repaired and checked by reading, not compiled, not run on hardware
//
#include <hip/hip_runtime.h>
#include <hip/hip_bf16.h>

typedef __attribute__((ext_vector_type(16))) __bf16 v16bf;
typedef __attribute__((ext_vector_type(8)))  __bf16 v8bf;
typedef __attribute__((ext_vector_type(8)))  float  v8f;

// ---------------- edge weight: ew[e] = edge_attr[e,:16] . fc1_w + fc1_b ----
__global__ void k_edge_w(const float* __restrict__ ea, const float* __restrict__ w,
                         const float* __restrict__ b, float* __restrict__ ew, int E) {
  int e = blockIdx.x * blockDim.x + threadIdx.x;
  if (e >= E) return;
  float s = b[0];
#pragma unroll
  for (int i = 0; i < 4; ++i) {
    float4 a = *(const float4*)(ea + (size_t)e * 16 + i * 4);
    s += a.x * w[i * 4 + 0] + a.y * w[i * 4 + 1] + a.z * w[i * 4 + 2] + a.w * w[i * 4 + 3];
  }
  ew[e] = s;
}

// ---------------- degree with self loops, then dis = rsqrt(deg) -----------
__global__ void k_deg_init(float* deg, int n) {
  int i = blockIdx.x * blockDim.x + threadIdx.x;
  if (i < n) deg[i] = 1.0f;  // self-loop weight 1
}

__global__ void k_deg_acc(const int* __restrict__ dst, const float* __restrict__ ew,
                          float* deg, int E) {
  int e = blockIdx.x * blockDim.x + threadIdx.x;
  if (e < E) atomicAdd(&deg[dst[e]], ew[e]);
}

__global__ void k_dis(float* deg, int n) {
  int i = blockIdx.x * blockDim.x + threadIdx.x;
  if (i >= n) return;
  float d = deg[i];
  deg[i] = d > 0.0f ? rsqrtf(fmaxf(d, 1e-12f)) : 0.0f;
}

// ---------------- fp32 -> bf16 conversion, optional fused ELU -------------
template <int DO_ELU>
__global__ void k_cvt_bf16(const float* __restrict__ x, __bf16* __restrict__ xb, size_t total) {
  size_t i = ((size_t)blockIdx.x * blockDim.x + threadIdx.x) * 2;
  if (i >= total) return;
  float2 v = *(const float2*)(x + i);
  if (DO_ELU) {
    v.x = v.x > 0.0f ? v.x : expm1f(v.x);
    v.y = v.y > 0.0f ? v.y : expm1f(v.y);
  }
  xb[i + 0] = (__bf16)v.x;
  xb[i + 1] = (__bf16)v.y;
}

// ---------------- W[K,NOUT] fp32 -> Wt[NOUT,K] bf16 (transpose) -----------
__global__ void k_wt_bf16(const float* __restrict__ W, __bf16* __restrict__ Wt,
                          int K, int NOUT) {
  int i = blockIdx.x * blockDim.x + threadIdx.x;
  if (i >= K * NOUT) return;
  int k = i / NOUT, c = i % NOUT;
  Wt[(size_t)c * K + k] = (__bf16)W[i];
}

// ---------------- WMMA GEMM + fused self-loop/bias epilogue ----------------
// H[n,NOUT] = Xb[n,K] @ Wt[NOUT,K]^T  (bf16 in, f32 accumulate)
// Y[n,NOUT] = H * dis[n]^2 + bias     (self-loop contribution, init for scatter)
// Block = 256 threads = 8 waves; each wave owns a 16x16 tile; K fully unrolled.
// REQUIRES n % 16 == 0 (holds: N = 50000 = 3125 * 16) -> no row guards.
template <int K, int NOUT>
__global__ void k_gemm_wmma(const __bf16* __restrict__ Xb, const __bf16* __restrict__ Wt,
                            const float* __restrict__ dis, const float* __restrict__ bias,
                            float* __restrict__ H, float* __restrict__ Y) {
  const int tid  = threadIdx.x;
  const int wave = tid >> 5;
  const int lane = tid & 31;
  const int r0 = blockIdx.x * 16;
  const int c0 = blockIdx.y * 128 + wave * 16;
  const int m  = lane & 15;  // row (A) / col (B,C,D) within tile
  const int hi = lane >> 4;  // lane-half select

  // A fragment (16-bit A 16x32 layout): lane m, VGPR0-3 = K[k0+8*hi .. +8),
  // VGPR4-7 = K[k0+16+8*hi .. +8)  -> two contiguous 16B chunks per k-step.
  const __bf16* arow = Xb + (size_t)(r0 + m) * K + hi * 8;
  // B fragment: lane col c0+m, same K order as A -> two contiguous 16B chunks per k-step.
  const __bf16* bcol = Wt + (size_t)(c0 + m) * K + hi * 8;

  v8f acc = {};
#pragma unroll
  for (int k0 = 0; k0 < K; k0 += 32) {
    v8bf alo = *(const v8bf*)(arow + k0);
    v8bf ahi = *(const v8bf*)(arow + k0 + 16);
    v16bf a = __builtin_shufflevector(alo, ahi, 0, 1, 2, 3, 4, 5, 6, 7,
                                      8, 9, 10, 11, 12, 13, 14, 15);
    v8bf blo = *(const v8bf*)(bcol + k0);
    v8bf bhi = *(const v8bf*)(bcol + k0 + 16);
    v16bf b = __builtin_shufflevector(blo, bhi, 0, 1, 2, 3, 4, 5, 6, 7,
                                      8, 9, 10, 11, 12, 13, 14, 15);
    acc = __builtin_amdgcn_wmma_f32_16x16x32_bf16(false, a, false, b,
                                                  (short)0, acc, false, false);
  }

  // C/D layout: VGPR r, lanes 0-15 -> M=r, lanes 16-31 -> M=r+8.
  // Straight-line stores: per row, 16 lanes write 64B contiguous in H and Y.
  const int col = c0 + m;
  const float bv = bias[col];
#pragma unroll
  for (int r = 0; r < 8; ++r) {
    const int row = r0 + hi * 8 + r;
    const float h = acc[r];
    const float s = dis[row];
    H[(size_t)row * NOUT + col] = h;
    Y[(size_t)row * NOUT + col] = h * s * s + bv;  // self-loop + bias
  }
}

// ---------------- edge gather-scale-scatter (atomic segment_sum) ----------
template <int C>
__global__ void k_scatter(const float* __restrict__ h, const int* __restrict__ src,
                          const int* __restrict__ dst, const float* __restrict__ ew,
                          const float* __restrict__ dis, float* __restrict__ y, int E) {
  const unsigned NC = C / 4;
  unsigned tid = blockIdx.x * blockDim.x + threadIdx.x;
  unsigned e = tid / NC;
  if (e >= (unsigned)E) return;
  int c = (int)(tid % NC) * 4;
  int s = src[e], d = dst[e];
  float coef = dis[s] * ew[e] * dis[d];
  float4 hv = *(const float4*)(h + (size_t)s * C + c);
  float* yp = y + (size_t)d * C + c;
  atomicAdd(yp + 0, hv.x * coef);
  atomicAdd(yp + 1, hv.y * coef);
  atomicAdd(yp + 2, hv.z * coef);
  atomicAdd(yp + 3, hv.w * coef);
}

// ---------------- ELU (alpha = 1), fp32 in-place ---------------------------
__global__ void k_elu(float* y, size_t total) {
  size_t i = (size_t)blockIdx.x * blockDim.x + threadIdx.x;
  if (i >= total) return;
  float v = y[i];
  y[i] = v > 0.0f ? v : expm1f(v);
}

// ---------------- global mean pool -----------------------------------------
__global__ void k_pool_zero(float* out, float* cnt) {
  int i = blockIdx.x * blockDim.x + threadIdx.x;
  if (i < 64 * 128) out[i] = 0.0f;
  if (i < 64) cnt[i] = 0.0f;
}

__global__ void k_pool_acc(const float* __restrict__ y, const int* __restrict__ batch,
                           float* out, float* cnt, int n) {
  int idx = blockIdx.x * blockDim.x + threadIdx.x;
  if (idx >= n * 32) return;
  int node = idx >> 5, c = (idx & 31) * 4;
  int g = batch[node];
  float4 v = *(const float4*)(y + (size_t)node * 128 + c);
  float* op = out + g * 128 + c;
  atomicAdd(op + 0, v.x);
  atomicAdd(op + 1, v.y);
  atomicAdd(op + 2, v.z);
  atomicAdd(op + 3, v.w);
  if (c == 0) atomicAdd(&cnt[g], 1.0f);
}

__global__ void k_pool_div(float* out, const float* cnt) {
  int i = blockIdx.x * blockDim.x + threadIdx.x;
  if (i >= 64 * 128) return;
  out[i] /= fmaxf(cnt[i >> 7], 1.0f);
}

extern "C" void kernel_launch(void* const* d_in, const int* in_sizes, int n_in,
                              void* d_out, int out_size, void* d_ws, size_t ws_size,
                              hipStream_t stream) {
  (void)n_in; (void)out_size; (void)ws_size;
  const float* x     = (const float*)d_in[0];
  const int*   eidx  = (const int*)d_in[1];
  const float* eattr = (const float*)d_in[2];
  const int*   batch = (const int*)d_in[3];
  const float* fc1w  = (const float*)d_in[4];
  const float* fc1b  = (const float*)d_in[5];
  const float* W1 = (const float*)d_in[6];  const float* b1 = (const float*)d_in[7];
  const float* W2 = (const float*)d_in[8];  const float* b2 = (const float*)d_in[9];
  const float* W3 = (const float*)d_in[10]; const float* b3 = (const float*)d_in[11];
  float* out = (float*)d_out;

  const int n = in_sizes[0] / 128;  // 50000 nodes (multiple of 16)
  const int E = in_sizes[1] / 2;    // 800000 edges
  const int* src = eidx;
  const int* dst = eidx + E;

  // ws layout (fp32 first, then bf16; all 32B-aligned given 256B-aligned base):
  // ew[E] | dis[n] | cnt[64] | bufA[n*256] | bufB[n*256] || Xb[n*256] | Wt1|Wt2|Wt3
  float* ws   = (float*)d_ws;
  float* ew   = ws;
  float* dis  = ew + E;
  float* cnt  = dis + n;
  float* bufA = cnt + 64;
  float* bufB = bufA + (size_t)n * 256;
  __bf16* Xb  = (__bf16*)(bufB + (size_t)n * 256);
  __bf16* Wt1 = Xb + (size_t)n * 256;         // [128][128]
  __bf16* Wt2 = Wt1 + 128 * 128;              // [256][128]
  __bf16* Wt3 = Wt2 + 128 * 256;              // [128][256]

  const int B = 256;
  const int mt = n / 16;

  // edge weights + symmetric norm
  k_edge_w  <<<(E + B - 1) / B, B, 0, stream>>>(eattr, fc1w, fc1b, ew, E);
  k_deg_init<<<(n + B - 1) / B, B, 0, stream>>>(dis, n);
  k_deg_acc <<<(E + B - 1) / B, B, 0, stream>>>(dst, ew, dis, E);
  k_dis     <<<(n + B - 1) / B, B, 0, stream>>>(dis, n);

  // bf16 operand prep
  k_wt_bf16<<<(128 * 128 + B - 1) / B, B, 0, stream>>>(W1, Wt1, 128, 128);
  k_wt_bf16<<<(128 * 256 + B - 1) / B, B, 0, stream>>>(W2, Wt2, 128, 256);
  k_wt_bf16<<<(256 * 128 + B - 1) / B, B, 0, stream>>>(W3, Wt3, 256, 128);
  k_cvt_bf16<0><<<((size_t)n * 64 + B - 1) / B, B, 0, stream>>>(x, Xb, (size_t)n * 128);

  // ---- layer 1: 128 -> 128 ----
  k_gemm_wmma<128, 128><<<dim3(mt, 1), B, 0, stream>>>(Xb, Wt1, dis, b1, bufA, bufB);
  k_scatter<128><<<((size_t)E * 32 + B - 1) / B, B, 0, stream>>>(bufA, src, dst, ew, dis, bufB, E);
  k_cvt_bf16<1><<<((size_t)n * 64 + B - 1) / B, B, 0, stream>>>(bufB, Xb, (size_t)n * 128);

  // ---- layer 2: 128 -> 256 ----
  k_gemm_wmma<128, 256><<<dim3(mt, 2), B, 0, stream>>>(Xb, Wt2, dis, b2, bufA, bufB);
  k_scatter<256><<<((size_t)E * 64 + B - 1) / B, B, 0, stream>>>(bufA, src, dst, ew, dis, bufB, E);
  k_cvt_bf16<1><<<((size_t)n * 128 + B - 1) / B, B, 0, stream>>>(bufB, Xb, (size_t)n * 256);

  // ---- layer 3: 256 -> 128 ----
  k_gemm_wmma<256, 128><<<dim3(mt, 1), B, 0, stream>>>(Xb, Wt3, dis, b3, bufA, bufB);
  k_scatter<128><<<((size_t)E * 32 + B - 1) / B, B, 0, stream>>>(bufA, src, dst, ew, dis, bufB, E);
  k_elu<<<((size_t)n * 128 + B - 1) / B, B, 0, stream>>>(bufB, (size_t)n * 128);

  // ---- global mean pool ----
  k_pool_zero<<<(64 * 128 + B - 1) / B, B, 0, stream>>>(out, cnt);
  k_pool_acc<<<((size_t)n * 32 + B - 1) / B, B, 0, stream>>>(bufB, batch, out, cnt, n);
  k_pool_div<<<(64 * 128 + B - 1) / B, B, 0, stream>>>(out, cnt);
}
